// CwRNN_34952443854946
// MI455X (gfx1250) — compile-verified, matrix operand order checked by reading, not run on hardware
//
#include <hip/hip_runtime.h>
#include <hip/hip_bf16.h>
#include <math.h>

// ---------------------------------------------------------------------------
// Clockwork RNN for MI455X (gfx1250), wave32 + WMMA bf16.
//   B=128, T=512, I=2, H=1024, NM=8, MS=128
// Strategy: batch-split persistent workgroups (Btile=16 -> 8 WGs), hidden
// state lives in LDS, recurrent weights pre-packed to WMMA B-layout bf16 in
// workspace (2 MB, L2-resident), main loop = v_wmma_f32_16x16x32_bf16.
// ---------------------------------------------------------------------------

typedef __attribute__((ext_vector_type(16))) __bf16 v16bf;
typedef __attribute__((ext_vector_type(8)))  __bf16 v8bf;
typedef __attribute__((ext_vector_type(8)))  float  v8f;

#define CW_B   128
#define CW_T   512
#define CW_I   2
#define CW_H   1024
#define CW_NM  8
#define CW_MS  128
#define CW_BT  16           // batch tile per workgroup
#define CW_PAD 8            // bf16 pad per LDS row (16 bytes -> bank spread)
#define CW_KT  32           // K tiles of 32 along H
#define CW_NT  64           // N tiles of 16 along H

// ---------------------------------------------------------------------------
// Prep: pack masked Whh^T (B[k][n] = mask(n,k) * W_hh[n][k]) into the per-lane
// WMMA B-operand layout, bf16.  Slot index: ((nt*32 + kt)*32 + lane)*16 + e.
// Within a lane's 16 elements: d = e/2, pos = e&1, hi = lane/16,
//   k = (d<4 ? 0 : 16) + hi*8 + (d&3)*2 + pos,  n = lane&15   (mirror of A).
// ---------------------------------------------------------------------------
__global__ __launch_bounds__(256) void cw_pack_whh(const float* __restrict__ W_hh,
                                                   __bf16* __restrict__ Bpack) {
    int gid = blockIdx.x * 256 + threadIdx.x;          // one bf16 element each
    int e    = gid & 15;
    int lane = (gid >> 4) & 31;
    int kt   = (gid >> 9) & 31;
    int nt   = gid >> 14;                               // 0..63
    int hi  = lane >> 4;
    int nl  = lane & 15;
    int d   = e >> 1;
    int pos = e & 1;
    int k   = ((d < 4) ? 0 : 16) + hi * 8 + (d & 3) * 2 + pos;
    int kg  = kt * 32 + k;
    int ng  = nt * 16 + nl;
    // mask_hh[n][k] = (module(k) >= module(n)), module = idx / 128
    float v = ((kg >> 7) >= (ng >> 7)) ? W_hh[ng * CW_H + kg] : 0.0f;
    Bpack[gid] = (__bf16)v;
}

// ---------------------------------------------------------------------------
// Main persistent kernel: one workgroup per 16-row batch tile, 8 waves.
// ---------------------------------------------------------------------------
__global__ __launch_bounds__(256) void cw_rnn_main(const float* __restrict__ x,
                                                   const float* __restrict__ W_ih,
                                                   const float* __restrict__ fc_w,
                                                   const float* __restrict__ fc_b,
                                                   const float* __restrict__ enc_w,
                                                   const __bf16* __restrict__ Bpack,
                                                   float* __restrict__ out) {
    __shared__ __bf16 h_lds[CW_BT][CW_H + CW_PAD];     // hidden state (bf16)
    __shared__ __bf16 cand_lds[CW_BT][CW_H + CW_PAD];  // per-step candidates
    __shared__ float  xs_t[CW_BT][CW_I];               // staged x_t
    __shared__ float  part[CW_BT * CW_I][8];           // fc partial sums

    const int tid  = threadIdx.x;
    const int wave = tid >> 5;
    const int lane = tid & 31;
    const int hi   = lane >> 4;
    const int nl   = lane & 15;
    const int b0   = blockIdx.x * CW_BT;

    // ---- h0 = x[:,0] @ enc_w.T  (I=2 -> two FMAs per element) ----
    for (int idx = tid; idx < CW_BT * CW_H; idx += 256) {
        int r = idx >> 10;
        int n = idx & (CW_H - 1);
        const float* xp = &x[(size_t)(b0 + r) * (CW_T + 1) * CW_I];
        float h0 = xp[0] * enc_w[n * CW_I + 0] + xp[1] * enc_w[n * CW_I + 1];
        h_lds[r][n] = (__bf16)h0;
    }

    for (int t = 0; t < CW_T; ++t) {
        // ---- stage x_t = x[:, t+1] ----
        if (tid < CW_BT * CW_I) {
            int r = tid >> 1, i = tid & 1;
            xs_t[r][i] = x[((size_t)(b0 + r) * (CW_T + 1) + (t + 1)) * CW_I + i];
        }
        __syncthreads();

        // ---- phase 1: cand = tanh(x_t W_ih^T + h Whh_m^T) for active modules
        int g = 0;
        for (int m = 0; m < CW_NM; ++m) {
            if ((t & ((1 << m) - 1)) != 0) continue;   // module m inactive
            for (int nt8 = 0; nt8 < 8; ++nt8, ++g) {
                if ((g & 7) != wave) continue;          // wave-uniform split
                int nt = m * 8 + nt8;                   // global 16-col tile
                v8f acc = {};
                // module m only reads hidden cols >= m*128 -> kt starts m*4
                for (int kt = m * 4; kt < CW_KT; ++kt) {
                    int kb = kt * 32;
                    union { v16bf v; v8bf h2[2]; } A;
                    A.h2[0] = *(const v8bf*)&h_lds[nl][kb + hi * 8];
                    A.h2[1] = *(const v8bf*)&h_lds[nl][kb + 16 + hi * 8];
                    const __bf16* bp =
                        &Bpack[(((nt * CW_KT) + kt) * 32 + lane) * 16];
                    __builtin_prefetch(bp + 32 * 16, 0, 1);   // next kt line
                    v16bf Bv = *(const v16bf*)bp;
                    acc = __builtin_amdgcn_wmma_f32_16x16x32_bf16(
                        false, A.v, false, Bv, (short)0, acc, false, false);
                }
                // epilogue: per-element bias + tanh, write candidate
                int Ng = nt * 16 + nl;                  // global output column
                float wi0 = W_ih[Ng * CW_I + 0];
                float wi1 = W_ih[Ng * CW_I + 1];
                for (int j = 0; j < 8; ++j) {
                    int M = j + hi * 8;                 // local batch row
                    float xb = xs_t[M][0] * wi0 + xs_t[M][1] * wi1;
                    cand_lds[M][Ng] = (__bf16)tanhf(acc[j] + xb);
                }
            }
        }
        __syncthreads();

        // ---- phase 2: h = where(active, cand, h) -> copy active modules ----
        for (int m = 0; m < CW_NM; ++m) {
            if ((t & ((1 << m) - 1)) != 0) continue;
            for (int idx = tid; idx < CW_BT * CW_MS; idx += 256) {
                int r = idx >> 7;
                int c = m * CW_MS + (idx & (CW_MS - 1));
                h_lds[r][c] = cand_lds[r][c];
            }
        }
        __syncthreads();

        // ---- fc: out[b,t,:] = h @ fc_w.T + fc_b   (32 outputs, 8 thr each)
        {
            int o = tid >> 3;                           // 0..31 : (b_local, i)
            int q = tid & 7;                            // 128-col slice
            int bl = o >> 1, i = o & 1;
            float s = 0.0f;
            int nbase = q * 128;
            for (int n = 0; n < 128; ++n) {
                s += (float)h_lds[bl][nbase + n] * fc_w[i * CW_H + nbase + n];
            }
            part[o][q] = s;
        }
        __syncthreads();
        if (tid < CW_BT * CW_I) {
            int bl = tid >> 1, i = tid & 1;
            float s = fc_b[i];
            for (int q = 0; q < 8; ++q) s += part[tid][q];
            out[((size_t)(b0 + bl) * CW_T + t) * CW_I + i] = s;
        }
        // next loop-top __syncthreads separates part reads from next writes
    }
}

// ---------------------------------------------------------------------------
extern "C" void kernel_launch(void* const* d_in, const int* in_sizes, int n_in,
                              void* d_out, int out_size, void* d_ws, size_t ws_size,
                              hipStream_t stream) {
    const float* x     = (const float*)d_in[0];
    const float* W_ih  = (const float*)d_in[1];
    const float* W_hh  = (const float*)d_in[2];
    const float* fc_w  = (const float*)d_in[3];
    const float* fc_b  = (const float*)d_in[4];
    const float* enc_w = (const float*)d_in[5];
    __bf16* Bpack = (__bf16*)d_ws;                      // 2 MB packed weights
    float* out = (float*)d_out;

    cw_pack_whh<<<(CW_H * CW_H) / 256, 256, 0, stream>>>(W_hh, Bpack);
    cw_rnn_main<<<CW_B / CW_BT, 256, 0, stream>>>(x, W_ih, fc_w, fc_b, enc_w,
                                                  Bpack, out);
}
